// RewardGNN_58411555226106
// MI455X (gfx1250) — matrix operand order repaired and checked by reading, not compiled, not run on hardware
//
#include <hip/hip_runtime.h>
#include <hip/hip_bf16.h>
#include <math.h>

// ---------------------------------------------------------------------------
// MI455X / gfx1250 implementation.
// Core compute: f16 WMMA 16x16x32 with f32 accumulation (all K % 32 == 0,
// all GEMM N % 64 == 0, all M % 32 == 0 in this model). Each wave computes a
// 32x64 output tile: 2 A fragments x 4 B fragments -> 8 WMMAs per 32-K chunk,
// with all 12 global_load_b128 of a chunk grouped ahead of the WMMA burst so
// they clause together and overlap matrix execution.
// Everything else (gathers, GRU pointwise, gumbel/graph, reductions) is
// exact f32 VALU work.
// ---------------------------------------------------------------------------

typedef __attribute__((ext_vector_type(16))) _Float16 v16h;
typedef __attribute__((ext_vector_type(8)))  float    v8f;

#define EPS_GS 1e-10f
#define TAU_GS 0.5f

union Frag16 {
  v16h v;
  uint4 q[2];
};

// ---------------------------------------------------------------------------
// Generic GEMM: C[M,N] = relu?( A[M,K](f16) * Bt[N,K](f16)^T + bias[N] )
// Requires M%32==0, N%64==0, K%32==0.  C (f32) and C16 (f16 mirror) optional.
// Fragment layouts per CDNA5 ISA 7.12.2:
//   A (16x32 f16): lane L -> row L&15; e<8: K = ah + e; e>=8: K = 16+ah+(e-8),
//                  ah = (L>=16)?8:0  -> two contiguous 8-half (16B) loads.
//   B (32x16 f16): lane L -> col L&15; same K mapping as A (bh = ah)
//                  -> two contiguous 16B loads from Bt[N,K].
//   C/D (16x16 f32): VGPR r -> row r + ((L>=16)?8:0), col L&15.
// ---------------------------------------------------------------------------
__global__ __launch_bounds__(256) void gemm_f16_wmma(
    const _Float16* __restrict__ A, const _Float16* __restrict__ Bt,
    const float* __restrict__ bias, float* __restrict__ C,
    _Float16* __restrict__ C16, int M, int N, int K, int relu)
{
  const int wave = threadIdx.x >> 5;
  const int lane = threadIdx.x & 31;
  const int tilesN = N >> 6;                       // 64-wide strips along N
  const int tile = blockIdx.x * 8 + wave;
  if (tile >= (M >> 5) * tilesN) return;           // uniform per wave
  const int tm = tile / tilesN;                    // 32-row tile index
  const int tn = tile % tilesN;
  const int l15 = lane & 15;
  const int hi  = lane >> 4;

  const _Float16* arow0 = A + (size_t)(tm * 32 + l15) * K + hi * 8;
  const _Float16* arow1 = arow0 + (size_t)16 * K;
  const _Float16* brow  = Bt + (size_t)(tn * 64 + l15) * K + hi * 8;
  const size_t bstride = (size_t)16 * K;

  v8f acc[2][4] = {};

  for (int k0 = 0; k0 < K; k0 += 32) {
    // ---- group all 12 16B loads of this chunk (clause + single wait) ----
    Frag16 a0, a1, b[4];
    a0.q[0] = *(const uint4*)(arow0 + k0);
    a0.q[1] = *(const uint4*)(arow0 + k0 + 16);
    a1.q[0] = *(const uint4*)(arow1 + k0);
    a1.q[1] = *(const uint4*)(arow1 + k0 + 16);
#pragma unroll
    for (int t = 0; t < 4; ++t) {
      const _Float16* bp = brow + (size_t)t * bstride + k0;
      b[t].q[0] = *(const uint4*)(bp);
      b[t].q[1] = *(const uint4*)(bp + 16);
    }
    // ---- 8 WMMAs on the staged fragments ----
#pragma unroll
    for (int t = 0; t < 4; ++t) {
      acc[0][t] = __builtin_amdgcn_wmma_f32_16x16x32_f16(
          false, a0.v, false, b[t].v, (short)0, acc[0][t], false, false);
      acc[1][t] = __builtin_amdgcn_wmma_f32_16x16x32_f16(
          false, a1.v, false, b[t].v, (short)0, acc[1][t], false, false);
    }
  }

#pragma unroll
  for (int m = 0; m < 2; ++m) {
    const int row0 = tm * 32 + m * 16 + hi * 8;
#pragma unroll
    for (int t = 0; t < 4; ++t) {
      const int col = tn * 64 + t * 16 + l15;
      const float bv = bias ? bias[col] : 0.f;
#pragma unroll
      for (int r = 0; r < 8; ++r) {
        float v = acc[m][t][r] + bv;
        if (relu) v = fmaxf(v, 0.f);
        const size_t idx = (size_t)(row0 + r) * N + col;
        if (C)   C[idx]   = v;
        if (C16) C16[idx] = (_Float16)v;
      }
    }
  }
}

// --------------------------- conversion kernels ----------------------------
__global__ void cvt_f32_f16(const float* __restrict__ s, _Float16* __restrict__ d,
                            long long count)
{
  long long id = (long long)blockIdx.x * blockDim.x + threadIdx.x;
  if (id < count) d[id] = (_Float16)s[id];
}

// W[K,N] f32 -> Wt[N,K] f16 (coalesced writes)
__global__ void cvt_w_f16t(const float* __restrict__ W, _Float16* __restrict__ Wt,
                           int K, int N)
{
  int id = blockIdx.x * blockDim.x + threadIdx.x;
  if (id >= K * N) return;
  int k = id % K, nn = id / K;
  Wt[id] = (_Float16)W[(size_t)k * N + nn];
}

// --------------------------- gather / concat kernels -----------------------
// pairx[(bt*n+i)*n+j] = [ demos[bt,i,:] | demos[bt,j,:] ]  (f16)
__global__ void build_pairx(_Float16* __restrict__ out, const float* __restrict__ demos,
                            int BT, int n, int f)
{
  long long total = (long long)BT * n * n * 2 * f;
  long long id = (long long)blockIdx.x * blockDim.x + threadIdx.x;
  if (id >= total) return;
  int c = (int)(id % (2 * f));
  long long r = id / (2 * f);
  int j = (int)(r % n); long long bti = r / n;
  int i = (int)(bti % n); long long bt = bti / n;
  float v = (c < f) ? demos[((bt * n) + i) * f + c]
                    : demos[((bt * n) + j) * f + (c - f)];
  out[id] = (_Float16)v;
}

// node_in[t][b*n+i] = [ h_node[(b*T+t)*n+i, :nf] | demos[b,t,i,:f] ]
__global__ void build_node_in(_Float16* __restrict__ out,
                              const _Float16* __restrict__ hnode16,
                              const float* __restrict__ demos,
                              int B, int T, int n, int nf, int f)
{
  const int W = nf + f;
  long long total = (long long)T * B * n * W;
  long long id = (long long)blockIdx.x * blockDim.x + threadIdx.x;
  if (id >= total) return;
  int c = (int)(id % W);
  long long row = id / W;
  int bi = (int)(row % ((long long)B * n));
  int t  = (int)(row / ((long long)B * n));
  int i = bi % n, b = bi / n;
  long long bt = (long long)b * T + t;
  out[id] = (c < nf) ? hnode16[(bt * n + i) * nf + c]
                     : (_Float16)demos[(bt * n + i) * f + (c - nf)];
}

// edge_in[t][(b*n+i)*n+j] = [ h_edge[bt,i,j,:nf] | demos[bt,i,:] | demos[bt,j,:] ]
__global__ void build_edge_in(_Float16* __restrict__ out,
                              const _Float16* __restrict__ hedge16,
                              const float* __restrict__ demos,
                              int B, int T, int n, int nf, int f)
{
  const int W = nf + 2 * f;
  const long long RE = (long long)B * n * n;
  long long total = (long long)T * RE * W;
  long long id = (long long)blockIdx.x * blockDim.x + threadIdx.x;
  if (id >= total) return;
  int c = (int)(id % W);
  long long row = id / W;
  long long r = row % RE;
  int t = (int)(row / RE);
  int j = (int)(r % n); long long bi = r / n;
  int i = (int)(bi % n); int b = (int)(bi / n);
  long long bt = (long long)b * T + t;
  _Float16 v;
  if (c < nf)           v = hedge16[((bt * n + i) * n + j) * nf + c];
  else if (c < nf + f)  v = (_Float16)demos[(bt * n + i) * f + (c - nf)];
  else                  v = (_Float16)demos[(bt * n + j) * f + (c - nf - f)];
  out[id] = v;
}

// Generic pairwise concat: out[(b*n+i)*n+j] = [s0 | s1 | s2], source row picked
// by mode: 0 -> (b*n+i), 1 -> (b*n+j), 2 -> (b*n+i)*n+j.
__global__ void gather_pair3(_Float16* __restrict__ out,
                             const _Float16* __restrict__ s0, int w0, int m0,
                             const _Float16* __restrict__ s1, int w1, int m1,
                             const _Float16* __restrict__ s2, int w2, int m2,
                             int B, int n)
{
  const int W = w0 + w1 + w2;
  long long total = (long long)B * n * n * W;
  long long id = (long long)blockIdx.x * blockDim.x + threadIdx.x;
  if (id >= total) return;
  int c = (int)(id % W);
  long long row = id / W;
  int j = (int)(row % n); long long bi = row / n;
  int i = (int)(bi % n);  int b = (int)(bi / n);
  const _Float16* s; int w, m;
  if (c < w0)            { s = s0; w = w0; m = m0; }
  else if (c < w0 + w1)  { s = s1; w = w1; m = m1; c -= w0; }
  else                   { s = s2; w = w2; m = m2; c -= (w0 + w1); }
  long long srow = (m == 0) ? ((long long)b * n + i)
                 : (m == 1) ? ((long long)b * n + j)
                            : row;
  out[id] = s[srow * w + c];
}

// Row-aligned concat of two f16 sources.
__global__ void gather_cat2rows(_Float16* __restrict__ out,
                                const _Float16* __restrict__ s0, int w0,
                                const _Float16* __restrict__ s1, int w1, int rows)
{
  const int W = w0 + w1;
  long long total = (long long)rows * W;
  long long id = (long long)blockIdx.x * blockDim.x + threadIdx.x;
  if (id >= total) return;
  int c = (int)(id % W);
  long long r = id / W;
  out[id] = (c < w0) ? s0[r * w0 + c] : s1[r * w1 + (c - w0)];
}

// --------------------------- GRU pointwise update --------------------------
// gx,gh: [R,3H] (gx already has bias). h: f32 state, h16: f16 mirror.
__global__ void gru_update(const float* __restrict__ gx, const float* __restrict__ gh,
                           float* __restrict__ h, _Float16* __restrict__ h16,
                           int R, int H)
{
  int id = blockIdx.x * blockDim.x + threadIdx.x;
  if (id >= R * H) return;
  int c = id % H, r = id / H;
  const float* gxr = gx + (size_t)r * 3 * H;
  const float* ghr = gh + (size_t)r * 3 * H;
  float z  = 1.f / (1.f + expf(-(gxr[c]         + ghr[c])));
  float rr = 1.f / (1.f + expf(-(gxr[H + c]     + ghr[H + c])));
  float cc = tanhf(gxr[2 * H + c] + rr * ghr[2 * H + c]);
  float hv = h[id];
  float hn = (1.f - z) * cc + z * hv;
  h[id] = hn;
  h16[id] = (_Float16)hn;
}

// --------------------------- affinity helpers ------------------------------
// agg16[b,i,h] = sum_j eh[b,i,j,h]
__global__ void aggsum_j(const float* __restrict__ eh, _Float16* __restrict__ agg16,
                         int B, int n, int hd)
{
  int id = blockIdx.x * blockDim.x + threadIdx.x;
  if (id >= B * n * hd) return;
  int h = id % hd, bi = id / hd;
  const float* base = eh + ((size_t)bi * n) * hd + h;
  float s = 0.f;
  for (int j = 0; j < n; ++j) s += base[(size_t)j * hd];
  agg16[id] = (_Float16)s;
}

// logits[r,k] = eh[r,:] . Wout[:,k] + bout[k], k<2
__global__ void logits_kernel(const float* __restrict__ eh, const float* __restrict__ Wout,
                              const float* __restrict__ bout, float* __restrict__ logits,
                              int rows, int K)
{
  int id = blockIdx.x * blockDim.x + threadIdx.x;
  if (id >= rows * 2) return;
  int k = id & 1, r = id >> 1;
  const float* e = eh + (size_t)r * K;
  float s = bout[k];
  for (int h = 0; h < K; ++h) s += e[h] * Wout[h * 2 + k];
  logits[id] = s;
}

// symmetrize + gumbel-softmax hard (straight-through) + mask + batch-mean
__global__ void graph_kernel(const float* __restrict__ logits, const float* __restrict__ u,
                             float* __restrict__ graph, int B, int n)
{
  int id = blockIdx.x * blockDim.x + threadIdx.x;
  if (id >= n * n) return;
  int i = id / n, j = id % n;
  float s0 = 0.f, s1 = 0.f;
  for (int b = 0; b < B; ++b) {
    long long rij = ((long long)b * n + i) * n + j;
    long long rji = ((long long)b * n + j) * n + i;
    float l0 = 0.5f * (logits[rij * 2 + 0] + logits[rji * 2 + 0]);
    float l1 = 0.5f * (logits[rij * 2 + 1] + logits[rji * 2 + 1]);
    float g0 = -logf(-logf(u[rij * 2 + 0] + EPS_GS) + EPS_GS);
    float g1 = -logf(-logf(u[rij * 2 + 1] + EPS_GS) + EPS_GS);
    float a0 = (l0 + g0) / TAU_GS, a1 = (l1 + g1) / TAU_GS;
    float m = fmaxf(a0, a1);
    float e0 = expf(a0 - m), e1 = expf(a1 - m);
    float inv = 1.f / (e0 + e1);
    float y0 = e0 * inv, y1 = e1 * inv;
    float h0 = (y0 >= y1) ? 1.f : 0.f;   // argmax tie -> index 0
    float h1 = 1.f - h0;
    float et0 = (h0 - y0) + y0;          // straight-through, same fp ordering
    float et1 = (h1 - y1) + y1;
    if (i == j) { et0 = 0.f; et1 = 0.f; }
    s0 += et0; s1 += et1;
  }
  graph[id * 2 + 0] = s0 / (float)B;
  graph[id * 2 + 1] = s1 / (float)B;
}

// aggm[b,i,h] = sum_{j,k} graph[i,j,k] * msg[k][b,i,j,h]   (writes f16 mirror)
__global__ void agg_msg(const float* __restrict__ graph, const float* __restrict__ msg,
                        _Float16* __restrict__ aggm16, int B, int n, int f)
{
  int id = blockIdx.x * blockDim.x + threadIdx.x;
  if (id >= B * n * f) return;
  int h = id % f, bi = id / f;
  int i = bi % n, b = bi / n;
  const size_t kstride = (size_t)B * n * n * f;
  float s = 0.f;
  for (int k = 0; k < 2; ++k) {
    const float* mk = msg + (size_t)k * kstride + (((size_t)b * n + i) * n) * f + h;
    const float* gk = graph + (size_t)i * n * 2 + k;
    for (int j = 0; j < n; ++j) s += gk[j * 2] * mk[(size_t)j * f];
  }
  aggm16[id] = (_Float16)s;
}

// out[b] = mean_i( hn[b,i,:] . Wr ) + br
__global__ void final_out(const float* __restrict__ hn, const float* __restrict__ Wr,
                          const float* __restrict__ br, float* __restrict__ out,
                          int B, int n, int f)
{
  int b = threadIdx.x;
  if (b >= B) return;
  float s = 0.f;
  for (int i = 0; i < n; ++i) {
    const float* row = hn + ((size_t)b * n + i) * f;
    float d = 0.f;
    for (int h = 0; h < f; ++h) d += row[h] * Wr[h];
    s += d;
  }
  out[b] = s / (float)n + br[0];
}

// ---------------------------------------------------------------------------
extern "C" void kernel_launch(void* const* d_in, const int* in_sizes, int n_in,
                              void* d_out, int out_size, void* d_ws, size_t ws_size,
                              hipStream_t stream)
{
  (void)in_sizes; (void)n_in; (void)out_size;
  const int B = 8, T = 16, n = 24, f = 128, nf = 64, hid = 192, H = 256;
  const int BT = B * T;          // 128
  const int RN = B * n;          // 192 node sequences
  const int RE = B * n * n;      // 4608 edge sequences
  const int MN = BT * n;         // 3072 node-frame rows
  const int ME = BT * n * n;     // 73728 edge-frame rows

  const float* node_rep = (const float*)d_in[0];
  const float* all_demos= (const float*)d_in[1];
  const float* u_noise  = (const float*)d_in[2];
  const float* enc_Wn1 = (const float*)d_in[3];  const float* enc_bn1 = (const float*)d_in[4];
  const float* enc_Wn2 = (const float*)d_in[5];  const float* enc_bn2 = (const float*)d_in[6];
  const float* enc_We1 = (const float*)d_in[7];  const float* enc_be1 = (const float*)d_in[8];
  const float* enc_We2 = (const float*)d_in[9];  const float* enc_be2 = (const float*)d_in[10];
  const float* grun_Wx = (const float*)d_in[11]; const float* grun_Wh = (const float*)d_in[12];
  const float* grun_b  = (const float*)d_in[13]; const float* grun_Wo = (const float*)d_in[14];
  const float* grun_bo = (const float*)d_in[15];
  const float* grue_Wx = (const float*)d_in[16]; const float* grue_Wh = (const float*)d_in[17];
  const float* grue_b  = (const float*)d_in[18]; const float* grue_Wo = (const float*)d_in[19];
  const float* grue_bo = (const float*)d_in[20];
  const float* aff_Wn  = (const float*)d_in[21]; const float* aff_bn  = (const float*)d_in[22];
  const float* aff_We  = (const float*)d_in[23]; const float* aff_be  = (const float*)d_in[24];
  const float* aff_Wpe = (const float*)d_in[25]; const float* aff_bpe = (const float*)d_in[26];
  const float* aff_Wpn = (const float*)d_in[27]; const float* aff_bpn = (const float*)d_in[28];
  const float* aff_Wout= (const float*)d_in[29]; const float* aff_bout= (const float*)d_in[30];
  const float* rw_Wm   = (const float*)d_in[31]; const float* rw_bm   = (const float*)d_in[32];
  const float* rw_Wu   = (const float*)d_in[33]; const float* rw_bu   = (const float*)d_in[34];
  const float* rw_Wr   = (const float*)d_in[35]; const float* rw_br   = (const float*)d_in[36];

  // ---- workspace bump allocator (256B aligned) ----
  size_t off = 0;
  auto alloc = [&](size_t bytes) -> char* {
    char* p = (char*)d_ws + off;
    off += (bytes + 255) & ~(size_t)255;
    return p;
  };
  auto h16 = [&](size_t elems) -> _Float16* { return (_Float16*)alloc(elems * 2); };
  auto f32 = [&](size_t elems) -> float*    { return (float*)alloc(elems * 4); };

  // f16 transposed weights [N,K]
  _Float16* wEncN1t = h16((size_t)192 * 128);
  _Float16* wEncN2t = h16((size_t)64  * 192);
  _Float16* wEncE1t = h16((size_t)192 * 256);
  _Float16* wEncE2t = h16((size_t)64  * 192);
  _Float16* wGrnXt  = h16((size_t)768 * 192);
  _Float16* wGrnHt  = h16((size_t)768 * 256);
  _Float16* wGrnOt  = h16((size_t)64  * 256);
  _Float16* wGreXt  = h16((size_t)768 * 320);
  _Float16* wGreHt  = h16((size_t)768 * 256);
  _Float16* wGreOt  = h16((size_t)64  * 256);
  _Float16* wAffNt  = h16((size_t)192 * 64);
  _Float16* wAffEt  = h16((size_t)192 * 192);
  _Float16* wAffPet = h16((size_t)2 * 192 * 576);
  _Float16* wAffPnt = h16((size_t)2 * 192 * 384);
  _Float16* wRwMt   = h16((size_t)2 * 128 * 256);
  _Float16* wRwUt   = h16((size_t)128 * 256);

  _Float16* x16     = h16((size_t)MN * f);       // all_demos f16 (node-enc A)
  _Float16* nrep16  = h16((size_t)RN * f);
  _Float16* hid1_16 = h16((size_t)MN * hid);
  _Float16* hnode16 = h16((size_t)MN * nf);

  // zone A: pairx(37.75MB)+ehid(28.3MB) later reused by edge_in (47.2MB)
  const size_t pairxBytes = (size_t)ME * 2 * f * 2;
  const size_t ehidBytes  = (size_t)ME * hid * 2;
  char* zoneA = alloc(pairxBytes + ehidBytes);
  _Float16* pairx16  = (_Float16*)zoneA;
  _Float16* ehid16   = (_Float16*)(zoneA + pairxBytes);
  _Float16* edgein16 = (_Float16*)zoneA;         // alias (after pairx/ehid dead)

  _Float16* hedge16  = h16((size_t)ME * nf);
  _Float16* nodein16 = h16((size_t)T * RN * (nf + f));
  float*    gxbuf    = f32((size_t)RE * 3 * H);  // shared by both GRUs
  float*    ghbuf    = f32((size_t)RE * 3 * H);
  float*    hnGru    = f32((size_t)RN * H);
  _Float16* hnGru16  = h16((size_t)RN * H);
  float*    heGru    = f32((size_t)RE * H);
  _Float16* heGru16  = h16((size_t)RE * H);
  _Float16* nodeAgg16= h16((size_t)RN * nf);
  _Float16* edgeAgg16= h16((size_t)RE * nf);
  _Float16* nh16     = h16((size_t)RN * hid);
  float*    ehF      = f32((size_t)RE * hid);
  _Float16* eh16     = h16((size_t)RE * hid);
  _Float16* catbuf   = h16((size_t)RE * 3 * hid);
  _Float16* agg16    = h16((size_t)RN * hid);
  _Float16* cat2buf  = h16((size_t)RN * 2 * hid);
  float*    logitsB  = f32((size_t)RE * 2);
  float*    graphB   = f32((size_t)n * n * 2);
  _Float16* pair2    = h16((size_t)RE * 2 * f);
  float*    msgB     = f32((size_t)2 * RE * f);
  _Float16* aggm16   = h16((size_t)RN * f);
  _Float16* catu     = h16((size_t)RN * 2 * f);
  float*    hnR      = f32((size_t)RN * f);

  if (off > ws_size) return;  // workspace too small -> do nothing safely

  // ---- launch helpers ----
  auto cvtT = [&](const float* W, _Float16* Wt, int K, int N) {
    int tot = K * N;
    cvt_w_f16t<<<(tot + 255) / 256, 256, 0, stream>>>(W, Wt, K, N);
  };
  auto gemm = [&](const _Float16* A, const _Float16* Bt, const float* bias,
                  float* C, _Float16* C16, int M, int N, int K, int relu) {
    int tiles = (M / 32) * (N / 64);
    gemm_f16_wmma<<<(tiles + 7) / 8, 256, 0, stream>>>(A, Bt, bias, C, C16, M, N, K, relu);
  };
  auto gridFor = [](long long total) { return dim3((unsigned)((total + 255) / 256)); };

  // ---- weight prep (f32 [K,N] -> f16 [N,K]) ----
  cvtT(enc_Wn1, wEncN1t, 128, 192);
  cvtT(enc_Wn2, wEncN2t, 192, 64);
  cvtT(enc_We1, wEncE1t, 256, 192);
  cvtT(enc_We2, wEncE2t, 192, 64);
  cvtT(grun_Wx, wGrnXt, 192, 768);
  cvtT(grun_Wh, wGrnHt, 256, 768);
  cvtT(grun_Wo, wGrnOt, 256, 64);
  cvtT(grue_Wx, wGreXt, 320, 768);
  cvtT(grue_Wh, wGreHt, 256, 768);
  cvtT(grue_Wo, wGreOt, 256, 64);
  cvtT(aff_Wn, wAffNt, 64, 192);
  cvtT(aff_We, wAffEt, 192, 192);
  for (int s = 0; s < 2; ++s) {
    cvtT(aff_Wpe + (size_t)s * 576 * 192, wAffPet + (size_t)s * 192 * 576, 576, 192);
    cvtT(aff_Wpn + (size_t)s * 384 * 192, wAffPnt + (size_t)s * 192 * 384, 384, 192);
  }
  for (int k = 0; k < 2; ++k)
    cvtT(rw_Wm + (size_t)k * 256 * 128, wRwMt + (size_t)k * 128 * 256, 256, 128);
  cvtT(rw_Wu, wRwUt, 256, 128);

  cvt_f32_f16<<<gridFor((long long)MN * f), 256, 0, stream>>>(all_demos, x16, (long long)MN * f);
  cvt_f32_f16<<<gridFor((long long)RN * f), 256, 0, stream>>>(node_rep, nrep16, (long long)RN * f);

  // ---- per-frame encoders ----
  gemm(x16, wEncN1t, enc_bn1, nullptr, hid1_16, MN, hid, f, 1);
  gemm(hid1_16, wEncN2t, enc_bn2, nullptr, hnode16, MN, nf, hid, 0);

  build_pairx<<<gridFor((long long)ME * 2 * f), 256, 0, stream>>>(pairx16, all_demos, BT, n, f);
  gemm(pairx16, wEncE1t, enc_be1, nullptr, ehid16, ME, hid, 2 * f, 1);
  gemm(ehid16, wEncE2t, enc_be2, nullptr, hedge16, ME, nf, hid, 0);

  // ---- build GRU input sequences (edge_in reuses zone A: pairx/ehid dead) ----
  build_node_in<<<gridFor((long long)T * RN * (nf + f)), 256, 0, stream>>>(
      nodein16, hnode16, all_demos, B, T, n, nf, f);
  build_edge_in<<<gridFor((long long)T * RE * (nf + 2 * f)), 256, 0, stream>>>(
      edgein16, hedge16, all_demos, B, T, n, nf, f);

  // ---- node GRU (R=192, H=256, T=16) ----
  hipMemsetAsync(hnGru, 0, (size_t)RN * H * 4, stream);
  hipMemsetAsync(hnGru16, 0, (size_t)RN * H * 2, stream);
  for (int t = 0; t < T; ++t) {
    gemm(nodein16 + (size_t)t * RN * (nf + f), wGrnXt, grun_b, gxbuf, nullptr, RN, 3 * H, nf + f, 0);
    gemm(hnGru16, wGrnHt, nullptr, ghbuf, nullptr, RN, 3 * H, H, 0);
    gru_update<<<gridFor((long long)RN * H), 256, 0, stream>>>(gxbuf, ghbuf, hnGru, hnGru16, RN, H);
  }
  gemm(hnGru16, wGrnOt, grun_bo, nullptr, nodeAgg16, RN, nf, H, 0);

  // ---- edge GRU (R=4608, H=256, T=16) ----
  hipMemsetAsync(heGru, 0, (size_t)RE * H * 4, stream);
  hipMemsetAsync(heGru16, 0, (size_t)RE * H * 2, stream);
  for (int t = 0; t < T; ++t) {
    gemm(edgein16 + (size_t)t * RE * (nf + 2 * f), wGreXt, grue_b, gxbuf, nullptr, RE, 3 * H, nf + 2 * f, 0);
    gemm(heGru16, wGreHt, nullptr, ghbuf, nullptr, RE, 3 * H, H, 0);
    gru_update<<<gridFor((long long)RE * H), 256, 0, stream>>>(gxbuf, ghbuf, heGru, heGru16, RE, H);
  }
  gemm(heGru16, wGreOt, grue_bo, nullptr, edgeAgg16, RE, nf, H, 0);

  // ---- affinity PropNet ----
  gemm(nodeAgg16, wAffNt, aff_bn, nullptr, nh16, RN, hid, nf, 1);
  gather_pair3<<<gridFor((long long)RE * 3 * nf), 256, 0, stream>>>(
      catbuf, nodeAgg16, nf, 0, nodeAgg16, nf, 1, edgeAgg16, nf, 2, B, n);
  gemm(catbuf, wAffEt, aff_be, ehF, eh16, RE, hid, 3 * nf, 1);

  for (int s = 0; s < 2; ++s) {
    gather_pair3<<<gridFor((long long)RE * 3 * hid), 256, 0, stream>>>(
        catbuf, nh16, hid, 0, nh16, hid, 1, eh16, hid, 2, B, n);
    gemm(catbuf, wAffPet + (size_t)s * 192 * 576, aff_bpe + (size_t)s * hid,
         ehF, eh16, RE, hid, 3 * hid, 1);
    aggsum_j<<<gridFor((long long)RN * hid), 256, 0, stream>>>(ehF, agg16, B, n, hid);
    gather_cat2rows<<<gridFor((long long)RN * 2 * hid), 256, 0, stream>>>(
        cat2buf, nh16, hid, agg16, hid, RN);
    gemm(cat2buf, wAffPnt + (size_t)s * 192 * 384, aff_bpn + (size_t)s * hid,
         nullptr, nh16, RN, hid, 2 * hid, 1);
  }

  logits_kernel<<<gridFor((long long)RE * 2), 256, 0, stream>>>(ehF, aff_Wout, aff_bout, logitsB, RE, hid);
  graph_kernel<<<gridFor((long long)n * n), 256, 0, stream>>>(logitsB, u_noise, graphB, B, n);

  // ---- reward GNN ----
  gather_pair3<<<gridFor((long long)RE * 2 * f), 256, 0, stream>>>(
      pair2, nrep16, f, 0, nrep16, f, 1, nullptr, 0, 2, B, n);
  for (int k = 0; k < 2; ++k)
    gemm(pair2, wRwMt + (size_t)k * 128 * 256, rw_bm + (size_t)k * f,
         msgB + (size_t)k * RE * f, nullptr, RE, f, 2 * f, 1);
  agg_msg<<<gridFor((long long)RN * f), 256, 0, stream>>>(graphB, msgB, aggm16, B, n, f);
  gather_cat2rows<<<gridFor((long long)RN * 2 * f), 256, 0, stream>>>(catu, nrep16, f, aggm16, f, RN);
  gemm(catu, wRwUt, rw_bu, hnR, nullptr, RN, f, 2 * f, 1);
  final_out<<<1, 32, 0, stream>>>(hnR, rw_Wr, rw_br, (float*)d_out, B, n, f);
}
